// TransformerBlock_43542378446883
// MI455X (gfx1250) — matrix operand order repaired and checked by reading, not compiled, not run on hardware
//
#include <hip/hip_runtime.h>
#include <cstdint>

#define EMB 1024
#define HEADS 16
#define HEAD_DIM 64
#define SEQ 2048
#define BATCH 2
#define FF_DIM 4096
#define TOKENS (BATCH*SEQ)

typedef __attribute__((ext_vector_type(16))) __bf16 v16bf;
typedef __attribute__((ext_vector_type(8)))  float  v8f;

union BF16x16 { uint4 q[2]; v16bf v; };

__device__ __forceinline__ unsigned short f2bf(float f) {
  union { float f; unsigned u; } cv; cv.f = f;
  unsigned u = cv.u;
  u += 0x7fffu + ((u >> 16) & 1u);   // round-to-nearest-even
  return (unsigned short)(u >> 16);
}

__device__ __forceinline__ float gelu_tanh(float x) {
  float x3 = x * x * x;
  return 0.5f * x * (1.0f + tanhf(0.7978845608028654f * (x + 0.044715f * x3)));
}

// ---------------------------------------------------------------------------
// Weight convert + transpose: W (K x N, f32) -> WT (N x K, bf16 bits)
// ---------------------------------------------------------------------------
__global__ __launch_bounds__(256)
void conv_transpose_bf16(const float* __restrict__ W, unsigned short* __restrict__ WT,
                         int K, int N) {
  int idx = blockIdx.x * 256 + threadIdx.x;
  int total = K * N;
  if (idx >= total) return;
  int k = idx / N;
  int n = idx - k * N;
  WT[(size_t)n * K + k] = f2bf(W[idx]);
}

// ---------------------------------------------------------------------------
// LayerNorm: f32 in -> bf16 out (one row of EMB per block)
// ---------------------------------------------------------------------------
__global__ __launch_bounds__(256)
void layernorm_bf16(const float* __restrict__ X, const float* __restrict__ g,
                    const float* __restrict__ beta, unsigned short* __restrict__ Out) {
  __shared__ float rs[256], rq[256];
  const int row = blockIdx.x;
  const int tid = threadIdx.x;
  const float* x = X + (size_t)row * EMB;
  float v[4], s = 0.f, s2 = 0.f;
#pragma unroll
  for (int i = 0; i < 4; ++i) {
    v[i] = x[tid + i * 256];
    s += v[i];
    s2 += v[i] * v[i];
  }
  rs[tid] = s; rq[tid] = s2;
  __syncthreads();
  for (int st = 128; st > 0; st >>= 1) {
    if (tid < st) { rs[tid] += rs[tid + st]; rq[tid] += rq[tid + st]; }
    __syncthreads();
  }
  const float mean = rs[0] * (1.0f / EMB);
  const float var  = rq[0] * (1.0f / EMB) - mean * mean;
  const float rstd = rsqrtf(var + 1e-5f);
#pragma unroll
  for (int i = 0; i < 4; ++i) {
    int c = tid + i * 256;
    Out[(size_t)row * EMB + c] = f2bf((v[i] - mean) * rstd * g[c] + beta[c]);
  }
}

// ---------------------------------------------------------------------------
// Epilogue helper
// ---------------------------------------------------------------------------
template<int MODE>
__device__ __forceinline__ void store_elem(void* __restrict__ OutP,
                                           const float* __restrict__ bias,
                                           const float* __restrict__ res,
                                           int row, int col, int N, float val) {
  if constexpr (MODE == 0) {
    ((unsigned short*)OutP)[(size_t)row * N + col] = f2bf(val);
  } else if constexpr (MODE == 1) {
    const int b = row >> 11, s = row & (SEQ - 1);
    const int h = col >> 6, d = col & (HEAD_DIM - 1);
    ((unsigned short*)OutP)[(((size_t)(b * HEADS + h)) * HEAD_DIM + d) * SEQ + s] = f2bf(val);
  } else if constexpr (MODE == 2) {
    ((float*)OutP)[(size_t)row * N + col] = val + bias[col] + res[(size_t)row * N + col];
  } else if constexpr (MODE == 3) {
    ((unsigned short*)OutP)[(size_t)row * N + col] = f2bf(gelu_tanh(val + bias[col]));
  } else { // MODE 4
    ((float*)OutP)[(size_t)row * N + col] = val + bias[col] + res[(size_t)row * N + col];
  }
}

// ---------------------------------------------------------------------------
// WMMA bf16 GEMM: C = A(MxK,bf16) * B(KxN), B stored transposed (NxK, bf16)
// Each wave computes a 32x64 tile (2 A-frags x 4 B-frags -> 8 accumulators,
// 8 WMMAs per 32-K step). Block = 8 waves as 4(M) x 2(N) -> 128 x 128 tile.
// Epilogue MODE: 0 = bf16 store
//                1 = bf16 transposed store into VT[B,H,D,S]
//                2 = f32 store: c + bias[col] + res[row*N+col]
//                3 = bf16 store: gelu(c + bias[col])
//                4 = f32 store: c + bias[col] + res[row*N+col]   (FF2 -> out)
// ---------------------------------------------------------------------------
template<int MODE>
__global__ __launch_bounds__(256)
void gemm_bf16_wmma(const unsigned short* __restrict__ A,
                    const unsigned short* __restrict__ BT,
                    void* __restrict__ OutP,
                    const float* __restrict__ bias,
                    const float* __restrict__ res,
                    int M, int N, int K) {
  const int lane  = threadIdx.x & 31;
  const int wM    = (threadIdx.x >> 5) & 3;   // 0..3
  const int wN    = threadIdx.x >> 7;         // 0..1
  const int ln    = lane & 15;
  const int half  = lane >> 4;                // wave32: lanes 16-31 second half
  const int tileM = blockIdx.y * 128 + wM * 32;
  const int tileN = blockIdx.x * 128 + wN * 64;

  v8f c0[4] = {}, c1[4] = {};

  // A-frag (16x32 bf16): lanes 0-15 hold K{0..15}, lanes 16-31 K{16..31}
  const unsigned short* arow0 = A + (size_t)(tileM + ln) * K + (half ? 16 : 0);
  const unsigned short* arow1 = arow0 + (size_t)16 * K;
  // B-frag (32x16 bf16 from transposed weights): per-lane 16 contiguous K values
  const unsigned short* brow[4];
#pragma unroll
  for (int t = 0; t < 4; ++t)
    brow[t] = BT + (size_t)(tileN + t * 16 + ln) * K + (half ? 16 : 0);

  for (int kk = 0; kk < K; kk += 32) {
    BF16x16 a0, a1;
    a0.q[0] = *(const uint4*)(arow0 + kk);
    a0.q[1] = *(const uint4*)(arow0 + kk + 8);
    a1.q[0] = *(const uint4*)(arow1 + kk);
    a1.q[1] = *(const uint4*)(arow1 + kk + 8);
    __builtin_prefetch(arow0 + kk + 64, 0, 3);   // global_prefetch_b8
    __builtin_prefetch(arow1 + kk + 64, 0, 3);
#pragma unroll
    for (int t = 0; t < 4; ++t) {
      BF16x16 b;
      b.q[0] = *(const uint4*)(brow[t] + kk);
      b.q[1] = *(const uint4*)(brow[t] + kk + 8);
      c0[t] = __builtin_amdgcn_wmma_f32_16x16x32_bf16(
          false, a0.v, false, b.v, (short)0, c0[t], false, false);
      c1[t] = __builtin_amdgcn_wmma_f32_16x16x32_bf16(
          false, a1.v, false, b.v, (short)0, c1[t], false, false);
    }
  }

  // C/D layout: element r of v8f holds row (r + half*8), column (tileN + t*16 + ln)
#pragma unroll
  for (int r = 0; r < 8; ++r) {
    const int row0 = tileM + r + half * 8;
    const int row1 = row0 + 16;
#pragma unroll
    for (int t = 0; t < 4; ++t) {
      const int col = tileN + t * 16 + ln;
      store_elem<MODE>(OutP, bias, res, row0, col, N, c0[t][r]);
      store_elem<MODE>(OutP, bias, res, row1, col, N, c1[t][r]);
    }
  }
}

// ---------------------------------------------------------------------------
// Flash attention: one wave per (batch, head, 16-query tile). Causal.
// Q,K: bf16 [TOKENS][EMB], VT: bf16 [B,H,D,S], Ctx out: bf16 [TOKENS][EMB]
// ---------------------------------------------------------------------------
__global__ __launch_bounds__(32)
void flash_attn(const unsigned short* __restrict__ Q,
                const unsigned short* __restrict__ Kmat,
                const unsigned short* __restrict__ VT,
                unsigned short* __restrict__ Ctx) {
  __shared__ __align__(16) unsigned short ldsP[16 * 32];  // P tile, 16 rows x 32 keys

  const int lane  = threadIdx.x & 31;
  const int ln    = lane & 15;
  const int half  = lane >> 4;
  const int qbase = blockIdx.x * 16;
  const int head  = blockIdx.y;
  const int b     = blockIdx.z;

  // Q A-frags for head_dim chunks d=[0,32) and d=[32,64)
  const unsigned short* qrow =
      Q + ((size_t)(b * SEQ + qbase + ln)) * EMB + head * HEAD_DIM + (half ? 16 : 0);
  BF16x16 aQ0, aQ1;
  aQ0.q[0] = *(const uint4*)(qrow);       aQ0.q[1] = *(const uint4*)(qrow + 8);
  aQ1.q[0] = *(const uint4*)(qrow + 32);  aQ1.q[1] = *(const uint4*)(qrow + 40);

  float mi[8], li[8];
  v8f o[4] = {};
#pragma unroll
  for (int r = 0; r < 8; ++r) { mi[r] = -1e30f; li[r] = 0.f; }

  for (int jj = 0; jj < qbase + 16; jj += 32) {
    // --- scores for 32 keys (two 16-key tiles) ---
    v8f s0 = {}, s1 = {};
    {
      const unsigned short* kr0 =
          Kmat + ((size_t)(b * SEQ + jj + ln)) * EMB + head * HEAD_DIM + (half ? 16 : 0);
      const unsigned short* kr1 = kr0 + (size_t)16 * EMB;
      BF16x16 bk;
      bk.q[0] = *(const uint4*)(kr0);      bk.q[1] = *(const uint4*)(kr0 + 8);
      s0 = __builtin_amdgcn_wmma_f32_16x16x32_bf16(false, aQ0.v, false, bk.v, (short)0, s0, false, false);
      bk.q[0] = *(const uint4*)(kr0 + 32); bk.q[1] = *(const uint4*)(kr0 + 40);
      s0 = __builtin_amdgcn_wmma_f32_16x16x32_bf16(false, aQ1.v, false, bk.v, (short)0, s0, false, false);
      bk.q[0] = *(const uint4*)(kr1);      bk.q[1] = *(const uint4*)(kr1 + 8);
      s1 = __builtin_amdgcn_wmma_f32_16x16x32_bf16(false, aQ0.v, false, bk.v, (short)0, s1, false, false);
      bk.q[0] = *(const uint4*)(kr1 + 32); bk.q[1] = *(const uint4*)(kr1 + 40);
      s1 = __builtin_amdgcn_wmma_f32_16x16x32_bf16(false, aQ1.v, false, bk.v, (short)0, s1, false, false);
    }

    // --- online softmax (row stats reduced across each 16-lane half) ---
    const float scale = 0.125f;  // 1/sqrt(64)
#pragma unroll
    for (int r = 0; r < 8; ++r) {
      const int qg = qbase + r + half * 8;
      float v0 = s0[r] * scale; if (jj + ln > qg)       v0 = -1e30f;
      float v1 = s1[r] * scale; if (jj + 16 + ln > qg)  v1 = -1e30f;
      float mx = fmaxf(v0, v1);
#pragma unroll
      for (int off = 8; off > 0; off >>= 1) mx = fmaxf(mx, __shfl_xor(mx, off, 32));
      const float mn = fmaxf(mi[r], mx);
      const float alpha = __expf(mi[r] - mn);
      const float p0 = __expf(v0 - mn);
      const float p1 = __expf(v1 - mn);
      float rs = p0 + p1;
#pragma unroll
      for (int off = 8; off > 0; off >>= 1) rs += __shfl_xor(rs, off, 32);
      li[r] = li[r] * alpha + rs;
      mi[r] = mn;
#pragma unroll
      for (int t = 0; t < 4; ++t) o[t][r] *= alpha;
      const int row16 = r + half * 8;
      ldsP[row16 * 32 + ln]      = f2bf(p0);
      ldsP[row16 * 32 + 16 + ln] = f2bf(p1);
    }

    asm volatile("s_wait_dscnt 0" ::: "memory");  // LDS stores visible to same-wave loads

    // --- P (C-layout) -> A-layout via LDS bounce ---
    BF16x16 aP;
    const unsigned short* lrow = ldsP + ln * 32 + (half ? 16 : 0);
    aP.q[0] = *(const uint4*)(lrow);
    aP.q[1] = *(const uint4*)(lrow + 8);

    // --- O += P * V ---
#pragma unroll
    for (int t = 0; t < 4; ++t) {
      const unsigned short* vr =
          VT + (((size_t)(b * HEADS + head)) * HEAD_DIM + t * 16 + ln) * SEQ + jj + (half ? 16 : 0);
      BF16x16 bv;
      bv.q[0] = *(const uint4*)(vr);
      bv.q[1] = *(const uint4*)(vr + 8);
      o[t] = __builtin_amdgcn_wmma_f32_16x16x32_bf16(false, aP.v, false, bv.v, (short)0, o[t], false, false);
    }
  }

  // --- normalize + write ctx (bf16 [token][EMB]) ---
#pragma unroll
  for (int r = 0; r < 8; ++r) {
    const float inv = 1.0f / li[r];
    const int row = qbase + r + half * 8;
#pragma unroll
    for (int t = 0; t < 4; ++t) {
      Ctx[((size_t)(b * SEQ + row)) * EMB + head * HEAD_DIM + t * 16 + ln] = f2bf(o[t][r] * inv);
    }
  }
}

// ---------------------------------------------------------------------------
// Host-side orchestration
// ---------------------------------------------------------------------------
extern "C" void kernel_launch(void* const* d_in, const int* in_sizes, int n_in,
                              void* d_out, int out_size, void* d_ws, size_t ws_size,
                              hipStream_t stream) {
  (void)in_sizes; (void)n_in; (void)out_size;
  const float* x     = (const float*)d_in[0];
  const float* Wq    = (const float*)d_in[1];
  const float* Wk    = (const float*)d_in[2];
  const float* Wv    = (const float*)d_in[3];
  const float* Wo    = (const float*)d_in[4];
  const float* bo    = (const float*)d_in[5];
  const float* ln1_g = (const float*)d_in[6];
  const float* ln1_b = (const float*)d_in[7];
  const float* ln2_g = (const float*)d_in[8];
  const float* ln2_b = (const float*)d_in[9];
  const float* W1    = (const float*)d_in[10];
  const float* b1    = (const float*)d_in[11];
  const float* W2    = (const float*)d_in[12];
  const float* b2    = (const float*)d_in[13];

  const size_t MB = 1u << 20;
  char* ws = (char*)d_ws;
  unsigned short* wqT = (unsigned short*)(ws + 0 * MB);    // 2MB
  unsigned short* wkT = (unsigned short*)(ws + 2 * MB);    // 2MB
  unsigned short* wvT = (unsigned short*)(ws + 4 * MB);    // 2MB
  unsigned short* woT = (unsigned short*)(ws + 6 * MB);    // 2MB
  unsigned short* w1T = (unsigned short*)(ws + 8 * MB);    // 8MB
  unsigned short* w2T = (unsigned short*)(ws + 16 * MB);   // 8MB
  unsigned short* hb  = (unsigned short*)(ws + 24 * MB);   // 8MB
  unsigned short* qb  = (unsigned short*)(ws + 32 * MB);   // 8MB
  unsigned short* kb  = (unsigned short*)(ws + 40 * MB);   // 8MB
  unsigned short* vtb = (unsigned short*)(ws + 48 * MB);   // 8MB
  unsigned short* ctx = (unsigned short*)(ws + 56 * MB);   // 8MB
  float*          x2  = (float*)(ws + 64 * MB);            // 16MB
  unsigned short* h2  = (unsigned short*)(ws + 80 * MB);   // 8MB
  unsigned short* a1  = (unsigned short*)(ws + 88 * MB);   // 32MB  (end 120MB)
  if (ws_size < 120 * MB) return;

  // 1) weights -> bf16 transposed
  conv_transpose_bf16<<<(EMB * EMB + 255) / 256, 256, 0, stream>>>(Wq, wqT, EMB, EMB);
  conv_transpose_bf16<<<(EMB * EMB + 255) / 256, 256, 0, stream>>>(Wk, wkT, EMB, EMB);
  conv_transpose_bf16<<<(EMB * EMB + 255) / 256, 256, 0, stream>>>(Wv, wvT, EMB, EMB);
  conv_transpose_bf16<<<(EMB * EMB + 255) / 256, 256, 0, stream>>>(Wo, woT, EMB, EMB);
  conv_transpose_bf16<<<(EMB * FF_DIM + 255) / 256, 256, 0, stream>>>(W1, w1T, EMB, FF_DIM);
  conv_transpose_bf16<<<(FF_DIM * EMB + 255) / 256, 256, 0, stream>>>(W2, w2T, FF_DIM, EMB);

  // 2) LN1
  layernorm_bf16<<<TOKENS, 256, 0, stream>>>(x, ln1_g, ln1_b, hb);

  // 3) Q/K/V projections (block tile 128x128)
  dim3 gEmb(EMB / 128, TOKENS / 128);
  gemm_bf16_wmma<0><<<gEmb, 256, 0, stream>>>(hb, wqT, qb, nullptr, nullptr, TOKENS, EMB, EMB);
  gemm_bf16_wmma<0><<<gEmb, 256, 0, stream>>>(hb, wkT, kb, nullptr, nullptr, TOKENS, EMB, EMB);
  gemm_bf16_wmma<1><<<gEmb, 256, 0, stream>>>(hb, wvT, vtb, nullptr, nullptr, TOKENS, EMB, EMB);

  // 4) causal flash attention
  dim3 gAtt(SEQ / 16, HEADS, BATCH);
  flash_attn<<<gAtt, 32, 0, stream>>>(qb, kb, vtb, ctx);

  // 5) output projection + residual
  gemm_bf16_wmma<2><<<gEmb, 256, 0, stream>>>(ctx, woT, x2, bo, x, TOKENS, EMB, EMB);

  // 6) LN2
  layernorm_bf16<<<TOKENS, 256, 0, stream>>>(x2, ln2_g, ln2_b, h2);

  // 7) FFN
  dim3 gFF(FF_DIM / 128, TOKENS / 128);
  gemm_bf16_wmma<3><<<gFF, 256, 0, stream>>>(h2, w1T, a1, b1, nullptr, TOKENS, FF_DIM, EMB);
  gemm_bf16_wmma<4><<<gEmb, 256, 0, stream>>>(a1, w2T, d_out, b2, x2, TOKENS, EMB, FF_DIM);
}
